// MoELayer_39694087749956
// MI455X (gfx1250) — matrix operand order repaired and checked by reading, not compiled, not run on hardware
//
#include <hip/hip_runtime.h>

// ---------------------------------------------------------------------------
// MoE layer for MI455X (gfx1250): routed top-2 grouped GEMM with bf16 WMMA.
//   B=4096, D_IN=1024, D_HID=4096, D_OUT=1024, E=8, TOP_K=2
// Grouped GEMM: 128x128x32 block tiles, 8 waves (4Mx2N), 8 WMMAs/wave/K-step,
// double-buffered LDS, packed b32 staging of the transposed weight tile.
// ---------------------------------------------------------------------------

#define NB    4096
#define DIN   1024
#define DHID  4096
#define DOUT  1024
#define NE    8
#define NROWS 8192   // B * TOP_K, exact

typedef __attribute__((ext_vector_type(16))) __bf16 v16bf;
typedef __attribute__((ext_vector_type(8)))  __bf16 v8bf;
typedef __attribute__((ext_vector_type(4)))  __bf16 v4bf;
typedef __attribute__((ext_vector_type(2)))  __bf16 v2bf;
typedef __attribute__((ext_vector_type(8)))  float  v8f;
typedef __attribute__((ext_vector_type(4)))  float  v4f;

// ---------------------------------------------------------------------------
// Utility: zero a float buffer (also used to clear the int/float ws header).
// ---------------------------------------------------------------------------
__global__ void zero_f32(float* __restrict__ p, int n) {
    int i = blockIdx.x * blockDim.x + threadIdx.x;
    if (i < n) p[i] = 0.0f;
}

// ---------------------------------------------------------------------------
// Gating: one wave32 per token. 8 dot products of length 1024, wave-reduce,
// top-2 + softmax over the pair, full softmax for the aux loss statistics.
// ---------------------------------------------------------------------------
__global__ __launch_bounds__(256)
void moe_gating(const float* __restrict__ x, const float* __restrict__ Wg,
                const float* __restrict__ bg,
                int* __restrict__ topki, float* __restrict__ topkw,
                int* __restrict__ cnt, float* __restrict__ usage) {
    __shared__ float s_usage[NE];
    const int tid  = threadIdx.x;
    const int lane = tid & 31;
    const int wid  = tid >> 5;
    if (tid < NE) s_usage[tid] = 0.0f;
    __syncthreads();

    const int t = blockIdx.x * 8 + wid;   // 8 waves per block, one token each

    float acc[NE];
#pragma unroll
    for (int e = 0; e < NE; ++e) acc[e] = 0.0f;

    const float* xr = x + (size_t)t * DIN;
    for (int i = lane; i < DIN; i += 32) {
        const float xv = xr[i];
        const float* wr = Wg + i * NE;   // Wg row-major [DIN][E]: 8 contiguous
#pragma unroll
        for (int e = 0; e < NE; ++e) acc[e] = fmaf(xv, wr[e], acc[e]);
    }
#pragma unroll
    for (int e = 0; e < NE; ++e) {
#pragma unroll
        for (int off = 16; off > 0; off >>= 1)
            acc[e] += __shfl_xor(acc[e], off, 32);
    }

    if (lane == 0) {
        float lg[NE];
#pragma unroll
        for (int e = 0; e < NE; ++e) lg[e] = acc[e] + bg[e];
        // top-1 (strict > keeps lowest index on ties, matching top_k)
        int i1 = 0; float v1 = lg[0];
#pragma unroll
        for (int e = 1; e < NE; ++e) if (lg[e] > v1) { v1 = lg[e]; i1 = e; }
        int i2 = (i1 == 0) ? 1 : 0; float v2 = lg[i2];
#pragma unroll
        for (int e = 0; e < NE; ++e)
            if (e != i1 && lg[e] > v2) { v2 = lg[e]; i2 = e; }
        // softmax over the selected pair (v1 is the max)
        const float ew = __expf(v2 - v1);
        const float w1 = 1.0f / (1.0f + ew);
        const float w2 = ew * w1;
        topki[2 * t]     = i1;  topki[2 * t + 1] = i2;
        topkw[2 * t]     = w1;  topkw[2 * t + 1] = w2;
        atomicAdd(&cnt[i1], 1);
        atomicAdd(&cnt[i2], 1);
        // full softmax for expert-usage statistics
        float s = 0.0f, g[NE];
#pragma unroll
        for (int e = 0; e < NE; ++e) { g[e] = __expf(lg[e] - v1); s += g[e]; }
        const float inv = 1.0f / s;
#pragma unroll
        for (int e = 0; e < NE; ++e) atomicAdd(&s_usage[e], g[e] * inv);
    }
    __syncthreads();
    if (tid < NE) atomicAdd(&usage[tid], s_usage[tid] * (1.0f / (float)NB));
}

// ---------------------------------------------------------------------------
// Exclusive scan over 8 expert counts (single thread, trivial).
// ---------------------------------------------------------------------------
__global__ void moe_scan(const int* __restrict__ cnt, int* __restrict__ offs) {
    if (threadIdx.x == 0) {
        int s = 0;
        for (int e = 0; e < NE; ++e) { offs[e] = s; s += cnt[e]; }
    }
}

// ---------------------------------------------------------------------------
// Build per-expert compacted (token, weight) row lists.
// ---------------------------------------------------------------------------
__global__ __launch_bounds__(256)
void moe_fill(const int* __restrict__ topki, const float* __restrict__ topkw,
              const int* __restrict__ offs, int* __restrict__ fill,
              int* __restrict__ rowtok, float* __restrict__ roww) {
    const int t = blockIdx.x * blockDim.x + threadIdx.x;
#pragma unroll
    for (int k = 0; k < 2; ++k) {
        const int e = topki[2 * t + k];
        const int p = offs[e] + atomicAdd(&fill[e], 1);
        rowtok[p] = t;
        roww[p]   = topkw[2 * t + k];
    }
}

// ---------------------------------------------------------------------------
// Gather token rows (f32) into the contiguous bf16 activation matrix Xe.
// One block per row, 4 elements per thread.
// ---------------------------------------------------------------------------
__global__ __launch_bounds__(256)
void moe_gather(const float* __restrict__ x, const int* __restrict__ rowtok,
                __bf16* __restrict__ Xe) {
    const int r   = blockIdx.x;
    const int tok = rowtok[r];
    const float* src = x + (size_t)tok * DIN;
    __bf16*      dst = Xe + (size_t)r   * DIN;
    const int c = threadIdx.x * 4;
    const v4f v = *(const v4f*)(src + c);
    v4bf o = { (__bf16)v.x, (__bf16)v.y, (__bf16)v.z, (__bf16)v.w };
    *(v4bf*)(dst + c) = o;
}

// ---------------------------------------------------------------------------
// Grouped GEMM with bf16 WMMA, f32 accumulation, double-buffered LDS.
//   PHASE 1:  H[rows, DHID] = relu(Xe[rows, DIN] @ W1[e] + b1[e])  (store bf16)
//   PHASE 2:  out[token]   += w_row * (H[rows, DHID] @ W2[e] + b2[e])
// Weight tiles are read as f32 from global (L2-resident), converted to bf16,
// and stored TRANSPOSED in LDS ([n][k]) so B fragments are contiguous b128
// loads. Staging packs a (k, k+1) bf16 pair per column -> ds_store_b32.
// ---------------------------------------------------------------------------
template <int PHASE>
__global__ __launch_bounds__(256)
void moe_gemm(const __bf16* __restrict__ A, const float* __restrict__ W,
              const float* __restrict__ bias,
              const int* __restrict__ cnt, const int* __restrict__ offs,
              __bf16* __restrict__ Hout,
              const int* __restrict__ rowtok, const float* __restrict__ roww,
              float* __restrict__ out) {
    constexpr int K = (PHASE == 1) ? DIN  : DHID;
    constexpr int N = (PHASE == 1) ? DHID : DOUT;
    constexpr int STEPS = K / 32;
    constexpr int BKP = 40;                  // 32 + 8 pad (keeps 16B alignment)

    __shared__ __bf16 sA[2][128][BKP];       // [buf][m][k]
    __shared__ __bf16 sB[2][128][BKP];       // [buf][n][k]  (transposed weights)

    const int e      = blockIdx.z;
    const int rows_e = cnt[e];
    const int m0     = blockIdx.y * 128;
    if (m0 >= rows_e) return;                // uniform per block: EXEC stays full
    const int row_start = offs[e] + m0;
    const int rows_in   = min(128, rows_e - m0);
    const int n_blk     = blockIdx.x * 128;

    const __bf16* Abase = A + (size_t)row_start * K;
    const float*  Bbase = W + (size_t)e * K * N + n_blk;

    const int tid  = threadIdx.x;
    const int lane = tid & 31, wid = tid >> 5;
    const int wm   = wid >> 1, wn = wid & 1;       // 4 x 2 wave grid
    const int half = lane >> 4, l16 = lane & 15;

    const int a_row = tid >> 1;                    // 128 rows, 2 threads/row
    const int a_seg = (tid & 1) * 16;              // 16 bf16 each
    const int b_kk  = (tid & 15) * 2;              // even k pair for this thread
    const int b_n0  = (tid >> 4) * 8;              // 8 columns per thread

    v8f acc[2][4] = {};

    // staging registers (next tile in flight while computing current tile)
    v8bf aR0, aR1;
    v4f  bR0, bR1, bR2, bR3;

    auto load_global = [&](int kb) {
        if (a_row < rows_in) {
            const __bf16* src = Abase + (size_t)a_row * K + kb * 32 + a_seg;
            aR0 = *(const v8bf*)(src);
            aR1 = *(const v8bf*)(src + 8);
        } else {
            aR0 = (v8bf){};
            aR1 = (v8bf){};
        }
        const float* bs = Bbase + (size_t)(kb * 32 + b_kk) * N + b_n0;
        bR0 = *(const v4f*)(bs);          // row k,   cols n0..n0+3
        bR1 = *(const v4f*)(bs + 4);      // row k,   cols n0+4..n0+7
        bR2 = *(const v4f*)(bs + N);      // row k+1, cols n0..n0+3
        bR3 = *(const v4f*)(bs + N + 4);  // row k+1, cols n0+4..n0+7
    };

    auto store_lds = [&](int buf) {
        *(v8bf*)&sA[buf][a_row][a_seg]     = aR0;
        *(v8bf*)&sA[buf][a_row][a_seg + 8] = aR1;
#pragma unroll
        for (int j = 0; j < 4; ++j) {
            v2bf p0 = { (__bf16)bR0[j], (__bf16)bR2[j] };   // col n0+j
            *(v2bf*)&sB[buf][b_n0 + j][b_kk] = p0;
            v2bf p1 = { (__bf16)bR1[j], (__bf16)bR3[j] };   // col n0+4+j
            *(v2bf*)&sB[buf][b_n0 + 4 + j][b_kk] = p1;
        }
    };

    load_global(0);
    store_lds(0);
    __syncthreads();

    for (int kb = 0; kb < STEPS; ++kb) {
        const int cur = kb & 1;
        if (kb + 1 < STEPS) load_global(kb + 1);

        // ---- B fragments: lane(half,n) holds col n, K = half*16 .. +15 ----
        v16bf bf[4];
#pragma unroll
        for (int tn = 0; tn < 4; ++tn) {
            const int n = wn * 64 + tn * 16 + l16;
            const v8bf lo = *(const v8bf*)&sB[cur][n][half * 16];
            const v8bf hi = *(const v8bf*)&sB[cur][n][half * 16 + 8];
            bf[tn] = __builtin_shufflevector(lo, hi,
                     0,1,2,3,4,5,6,7,8,9,10,11,12,13,14,15);
        }
        // ---- A fragments: lane(half,m) holds row m, K = half*16 .. +15 ----
        v16bf af[2];
#pragma unroll
        for (int tm = 0; tm < 2; ++tm) {
            const int m = wm * 32 + tm * 16 + l16;
            const v8bf lo = *(const v8bf*)&sA[cur][m][half * 16];
            const v8bf hi = *(const v8bf*)&sA[cur][m][half * 16 + 8];
            af[tm] = __builtin_shufflevector(lo, hi,
                     0,1,2,3,4,5,6,7,8,9,10,11,12,13,14,15);
        }
        // ---- 8 WMMAs per wave per K-step (full EXEC) ----
#pragma unroll
        for (int tm = 0; tm < 2; ++tm)
#pragma unroll
            for (int tn = 0; tn < 4; ++tn)
                acc[tm][tn] = __builtin_amdgcn_wmma_f32_16x16x32_bf16(
                    false, af[tm], false, bf[tn],
                    (short)0, acc[tm][tn], false, false);

        if (kb + 1 < STEPS) {
            store_lds((kb + 1) & 1);   // other buffer: no race with current reads
            __syncthreads();
        }
    }

    // ---- epilogue: C layout lane(half, n=l16), VGPR r -> row half*8+r ----
#pragma unroll
    for (int tn = 0; tn < 4; ++tn) {
        const int col = n_blk + wn * 64 + tn * 16 + l16;
        const float bv = bias[e * N + col];
#pragma unroll
        for (int tm = 0; tm < 2; ++tm) {
#pragma unroll
            for (int r = 0; r < 8; ++r) {
                const int ml = wm * 32 + tm * 16 + half * 8 + r;
                if (ml < rows_in) {
                    float v = acc[tm][tn][r] + bv;
                    if (PHASE == 1) {
                        v = fmaxf(v, 0.0f);
                        Hout[(size_t)(row_start + ml) * DHID + col] = (__bf16)v;
                    } else {
                        const int gr = row_start + ml;
                        // exactly 2 adds per output element -> deterministic
                        unsafeAtomicAdd(&out[(size_t)rowtok[gr] * DOUT + col],
                                        roww[gr] * v);
                    }
                }
            }
        }
    }
}

// ---------------------------------------------------------------------------
// Load-balance loss: E * sum(mean_gates^2)
// ---------------------------------------------------------------------------
__global__ void moe_finalize(const float* __restrict__ usage,
                             float* __restrict__ out_loss) {
    if (threadIdx.x == 0) {
        float s = 0.0f;
        for (int e = 0; e < NE; ++e) s += usage[e] * usage[e];
        *out_loss = (float)NE * s;
    }
}

// ---------------------------------------------------------------------------
// Host-side orchestration.
// Workspace layout (bytes):
//   [0,32)    cnt[8]      [32,64)  fill[8]    [64,96) offs[8]   [96,128) usage[8]
//   [256,+32K)    topki[B*2]      (+32K) topkw[B*2]
//   (+64K)        rowtok[8192]    (+96K) roww[8192]
//   (+128K)       Xe  bf16 [8192 x 1024]   (16 MB)
//   (+128K+16MB)  H   bf16 [8192 x 4096]   (64 MB)
// Total ~80.2 MB of d_ws.
// ---------------------------------------------------------------------------
extern "C" void kernel_launch(void* const* d_in, const int* in_sizes, int n_in,
                              void* d_out, int out_size, void* d_ws, size_t ws_size,
                              hipStream_t stream) {
    const float* x  = (const float*)d_in[0];
    const float* Wg = (const float*)d_in[1];
    const float* bg = (const float*)d_in[2];
    const float* W1 = (const float*)d_in[3];
    const float* b1 = (const float*)d_in[4];
    const float* W2 = (const float*)d_in[5];
    const float* b2 = (const float*)d_in[6];
    float* out = (float*)d_out;

    char* ws = (char*)d_ws;
    int*    cnt    = (int*)  (ws + 0);
    int*    fill   = (int*)  (ws + 32);
    int*    offs   = (int*)  (ws + 64);
    float*  usage  = (float*)(ws + 96);
    int*    topki  = (int*)  (ws + 256);
    float*  topkw  = (float*)(ws + 256 + 1 * 32768);
    int*    rowtok = (int*)  (ws + 256 + 2 * 32768);
    float*  roww   = (float*)(ws + 256 + 3 * 32768);
    __bf16* Xe     = (__bf16*)(ws + 256 + 4 * 32768);
    __bf16* H      = (__bf16*)(ws + 256 + 4 * 32768 + (size_t)NROWS * DIN * 2);

    // 1) clear output accumulator + workspace header (counts/fills/usage)
    zero_f32<<<(NB * DOUT) / 256, 256, 0, stream>>>(out, NB * DOUT);
    zero_f32<<<1, 64, 0, stream>>>((float*)ws, 32);

    // 2) gating + routing statistics
    moe_gating<<<NB / 8, 256, 0, stream>>>(x, Wg, bg, topki, topkw, cnt, usage);

    // 3) per-expert offsets, row compaction, activation gather (f32 -> bf16)
    moe_scan<<<1, 1, 0, stream>>>(cnt, offs);
    moe_fill<<<NB / 256, 256, 0, stream>>>(topki, topkw, offs, fill, rowtok, roww);
    moe_gather<<<NROWS, 256, 0, stream>>>(x, rowtok, Xe);

    // 4) grouped GEMM 1: H = relu(Xe @ W1[e] + b1[e])
    moe_gemm<1><<<dim3(DHID / 128, NB / 128, NE), 256, 0, stream>>>(
        Xe, W1, b1, cnt, offs, H, nullptr, nullptr, nullptr);

    // 5) grouped GEMM 2: out[token] += w * (H @ W2[e] + b2[e])
    moe_gemm<2><<<dim3(DOUT / 128, NB / 128, NE), 256, 0, stream>>>(
        H, W2, b2, cnt, offs, nullptr, rowtok, roww, out);

    // 6) auxiliary load-balance loss (last element of d_out)
    moe_finalize<<<1, 1, 0, stream>>>(usage, out + (size_t)NB * DOUT);
}
